// EGNN_85117661872359
// MI455X (gfx1250) — matrix operand order repaired and checked by reading, not compiled, not run on hardware
//
#include <hip/hip_runtime.h>
#include <hip/hip_bf16.h>

// ---------------------------------------------------------------------------
// EGCN (3-layer) on MI455X / gfx1250.
//  * node linear GEMM -> v_wmma_f32_16x16x32_bf16 (fp32 accum), A tile staged
//    into LDS by the Tensor Data Mover (tensor_load_to_lds + s_wait_tensorcnt),
//    2 M-tiles per wave so B (weights) registers are reused across 2 WMMAs.
//  * edge encoder + gather + relu + scale + scatter fused in ONE kernel: no
//    [E,128] intermediate touches HBM; hx (51MB) + agg (51MB) stay in 192MB L2.
//  * BN stats via block-partial sums + float atomics on tiny [2*128] arrays.
// ---------------------------------------------------------------------------

typedef __attribute__((ext_vector_type(16))) __bf16 v16bf;
typedef __attribute__((ext_vector_type(8)))  float  v8f;
typedef __attribute__((ext_vector_type(4)))  unsigned int v4u;
typedef __attribute__((ext_vector_type(8)))  unsigned int v8u;

#define HCH 128

// ---------------- small elementwise kernels ----------------

__global__ void init_degs_k(float* __restrict__ degs, int n) {
    int i = blockIdx.x * 256 + threadIdx.x;
    if (i < n) degs[i] = 1.0f;                 // (+1.0 of reference)
}

__global__ void deg_count_k(const int* __restrict__ src, float* __restrict__ degs, int e) {
    int i = blockIdx.x * 256 + threadIdx.x;
    if (i < e) atomicAdd(&degs[src[i]], 1.0f);
}

__global__ void norm_k(const float* __restrict__ degs, float* __restrict__ nrm, int n) {
    int i = blockIdx.x * 256 + threadIdx.x;
    if (i < n) nrm[i] = rsqrtf(degs[i]);       // degs^-0.5
}

__global__ void h_init_k(const int* __restrict__ x, const float* __restrict__ node_emb,
                         float* __restrict__ h, int n) {
    int i = blockIdx.x * 256 + threadIdx.x;    // i over N*H
    if (i >= n) return;
    int node = i >> 7, c = i & (HCH - 1);
    h[i] = node_emb[x[node] * HCH + c];
}

__global__ void f32_to_bf16_k(const float* __restrict__ src, __bf16* __restrict__ dst, int n) {
    int i = blockIdx.x * 256 + threadIdx.x;
    if (i < n) dst[i] = (__bf16)src[i];
}

// ---------------- WMMA node-linear GEMM: hx = h @ W + b ----------------
// Block = 256 threads (8 waves). Block computes a 32-row x 128-col tile of hx.
// Wave w computes two 16x16 sub-tiles at columns [16w, 16w+16) (rows 0-15 and
// 16-31 of the block tile), reusing the B registers for both WMMAs.
// A (32x128 bf16, 8KB) is staged in LDS by the TDM (tensor_load_to_lds).
// Per-lane A layout per ISA 7.12.2:
//   lane: M = lane&15, half = lane>>4
//   vector elems 0..7  : K = kk*32 +      half*8 + 0..7   (contiguous)
//   vector elems 8..15 : K = kk*32 + 16 + half*8 + 0..7   (contiguous)
// B (32x16 bf16) per-lane: N = tileCol + (lane&15), elem s at the same K as A's elem s.
// C/D (16x16 f32): row = r + 8*(lane>>4), col = lane&15.
union BF16x16 { v16bf v; uint4 q[2]; };

__global__ void __launch_bounds__(256)
node_linear_wmma_k(const __bf16* __restrict__ hbf,   // [N,128] bf16 row-major
                   const __bf16* __restrict__ wbf,   // [128,128] bf16 row-major
                   const float*  __restrict__ bias,  // [128]
                   float* __restrict__ hx)           // [N,128] f32
{
    __shared__ __align__(16) __bf16 sA[32 * HCH];    // 8 KB tile of A
    const int t       = threadIdx.x;
    const int rowBase = blockIdx.x * 32;             // N = 100000 = 3125*32 exact

    if (t < 32) {
        // Tensor Data Mover: contiguous 8KB (1024 x 8B) global -> LDS copy.
        // D# group0: count=1 | lds_addr[63:32] | global_addr[120:64] | type=2.
        // Flat pointers to LDS keep the LDS byte offset in the low 32 bits.
        const unsigned lds_off = (unsigned)(unsigned long long)(&sA[0]);
        const unsigned long long ga =
            (unsigned long long)(const void*)(hbf + (size_t)rowBase * HCH);
        v4u g0;
        g0[0] = 1u;                                        // count = 1 valid D#
        g0[1] = lds_off;                                   // lds_addr
        g0[2] = (unsigned)(ga & 0xffffffffull);            // global_addr[31:0]
        g0[3] = (unsigned)((ga >> 32) & 0x1ffffffull)      // global_addr[56:32]
              | (2u << 30);                                // type = 2 ("image")
        // D# group1: data_size=8B; tensor_dim0=tile_dim0=stride=1024; 1 row.
        v8u g1;
        g1[0] = 3u << 16;        // [17:16] data_size = 3 (8 bytes)
        g1[1] = 1024u << 16;     // [79:48] tensor_dim0 = 1024 (low half)
        g1[2] = 1u << 16;        // [111:80] tensor_dim1 = 1
        g1[3] = 1024u << 16;     // [127:112] tile_dim0 = 1024
        g1[4] = 1u;              // [143:128] tile_dim1 = 1
        g1[5] = 1024u;           // [207:160] tensor_dim0_stride = 1024
        g1[6] = 0u;
        g1[7] = 0u;
        asm volatile("tensor_load_to_lds %0, %1" :: "s"(g0), "s"(g1) : "memory");
        __builtin_amdgcn_s_wait_tensorcnt(0);
    }
    __syncthreads();

    const int wave    = t >> 5;          // 0..7
    const int lane    = t & 31;
    const int tileCol = wave * 16;
    const int m       = lane & 15;
    const int half    = lane >> 4;

    v8f c0 = {}, c1 = {};
#pragma unroll
    for (int kk = 0; kk < 4; ++kk) {     // K = 128 in 4 steps of 32
        BF16x16 b;
        const __bf16* brow = wbf + (size_t)(kk * 32) * HCH + tileCol + m;
#pragma unroll
        for (int s = 0; s < 16; ++s)                           // K = table order, N = lane&15
            b.v[s] = brow[(size_t)(s < 8 ? half * 8 + s : 16 + half * 8 + (s - 8)) * HCH];
        BF16x16 a0, a1;
        const __bf16* arow0 = sA + m * HCH + kk * 32 + half * 8;
        a0.q[0] = *reinterpret_cast<const uint4*>(arow0);      // K_l = half*8+0..7
        a0.q[1] = *reinterpret_cast<const uint4*>(arow0 + 16); // K_l = 16+half*8+0..7
        const __bf16* arow1 = arow0 + 16 * HCH;                // rows 16..31
        a1.q[0] = *reinterpret_cast<const uint4*>(arow1);
        a1.q[1] = *reinterpret_cast<const uint4*>(arow1 + 16);
        c0 = __builtin_amdgcn_wmma_f32_16x16x32_bf16(
                false, a0.v, false, b.v, (short)0, c0, false, false);
        c1 = __builtin_amdgcn_wmma_f32_16x16x32_bf16(
                false, a1.v, false, b.v, (short)0, c1, false, false);
    }

    const int   col  = tileCol + m;      // C/D: N = lane&15
    const float bval = bias[col];
#pragma unroll
    for (int r = 0; r < 8; ++r) {
        const int row = rowBase + r + (half ? 8 : 0);
        hx[(size_t)row * HCH + col]        = c0[r] + bval;
        hx[(size_t)(row + 16) * HCH + col] = c1[r] + bval;
    }
}

// ---------------- fused edge kernel ----------------
// One (edge, channel) per thread; 2 edges per 256-thread block.
// m = enorm * relu(hx[src] + ex@edge_w + edge_b); atomicAdd into agg[dst].
__global__ void __launch_bounds__(256)
edge_fused_k(const float* __restrict__ ex, const int* __restrict__ src,
             const int* __restrict__ dst, const float* __restrict__ nrm,
             const float* __restrict__ hx, const float* __restrict__ ew,  // [7,128]
             const float* __restrict__ eb, float* __restrict__ agg)
{
    __shared__ float sEx[2][8];
    const int le = threadIdx.x >> 7;               // local edge 0/1
    const int c  = threadIdx.x & (HCH - 1);
    const int e  = blockIdx.x * 2 + le;            // E = 1.6M, divisible by 2
    if (c < 7) sEx[le][c] = ex[(size_t)e * 7 + c];
    __syncthreads();

    const int   s  = src[e], d = dst[e];
    const float en = nrm[s] * nrm[d];
    float acc = eb[c];
#pragma unroll
    for (int k = 0; k < 7; ++k) acc += sEx[le][k] * ew[k * HCH + c];
    float v = hx[(size_t)s * HCH + c] + acc;
    v = fmaxf(v, 0.0f) * en;
    atomicAdd(&agg[(size_t)d * HCH + c], v);
}

// ---------------- combine + BN statistics ----------------
// h = agg + relu(hx + root)/degs ; accumulate per-channel sum & sumsq.
__global__ void __launch_bounds__(128)
combine_stats_k(const float* __restrict__ agg, const float* __restrict__ hx,
                const float* __restrict__ root, const float* __restrict__ degs,
                float* __restrict__ h, float* __restrict__ stats, int n)
{
    const int c    = threadIdx.x;                  // channel
    const int row0 = blockIdx.x * 128;
    const float rc = root[c];
    float s = 0.0f, s2 = 0.0f;
    for (int r = 0; r < 128; ++r) {
        const int row = row0 + r;
        if (row >= n) break;
        const size_t idx = (size_t)row * HCH + c;
        const float v = agg[idx] + fmaxf(hx[idx] + rc, 0.0f) / degs[row];
        h[idx] = v;
        s += v; s2 += v * v;
    }
    atomicAdd(&stats[c], s);
    atomicAdd(&stats[HCH + c], s2);
}

__global__ void bn_apply_k(float* __restrict__ h, const float* __restrict__ stats,
                           const float* __restrict__ g, const float* __restrict__ b,
                           int do_relu, int total, float invN)
{
    const int i = blockIdx.x * 256 + threadIdx.x;
    if (i >= total) return;
    const int c   = i & (HCH - 1);
    const float mu  = stats[c] * invN;
    const float var = stats[HCH + c] * invN - mu * mu;
    float v = (h[i] - mu) * rsqrtf(var + 1e-5f) * g[c] + b[c];
    if (do_relu) v = fmaxf(v, 0.0f);
    h[i] = v;
}

// ---------------- mean pool + output projection ----------------
__global__ void __launch_bounds__(128)
pool_sum_k(const float* __restrict__ h, float* __restrict__ colsum, int n)
{
    const int c    = threadIdx.x;
    const int row0 = blockIdx.x * 128;
    float s = 0.0f;
    for (int r = 0; r < 128; ++r) {
        const int row = row0 + r;
        if (row >= n) break;
        s += h[(size_t)row * HCH + c];
    }
    atomicAdd(&colsum[c], s);
}

__global__ void __launch_bounds__(128)
out_proj_k(const float* __restrict__ colsum, const float* __restrict__ ow,  // [128,128]
           const float* __restrict__ ob, float* __restrict__ out, float invN)
{
    const int c = threadIdx.x;
    float acc = ob[c];
    for (int k = 0; k < HCH; ++k)
        acc += (colsum[k] * invN) * ow[k * HCH + c];
    out[c] = acc;
}

// ---------------- host launcher ----------------
extern "C" void kernel_launch(void* const* d_in, const int* in_sizes, int n_in,
                              void* d_out, int out_size, void* d_ws, size_t ws_size,
                              hipStream_t stream)
{
    const int N = in_sizes[0];       // 100000
    const int E = in_sizes[2];       // 1600000
    const int H = HCH, L = 3;

    const int*   x        = (const int*)  d_in[0];
    const float* ex       = (const float*)d_in[1];
    const int*   src      = (const int*)  d_in[2];
    const int*   dst      = (const int*)  d_in[3];
    const float* node_emb = (const float*)d_in[4];
    const float* lin_w    = (const float*)d_in[5];
    const float* lin_b    = (const float*)d_in[6];
    const float* root_emb = (const float*)d_in[7];
    const float* edge_w   = (const float*)d_in[8];
    const float* edge_b   = (const float*)d_in[9];
    const float* bn_g     = (const float*)d_in[10];
    const float* bn_b     = (const float*)d_in[11];
    const float* out_w    = (const float*)d_in[12];
    const float* out_b    = (const float*)d_in[13];
    float*       out      = (float*)d_out;

    // workspace carve (ws assumed 256B-aligned base)
    char* ws = (char*)d_ws;
    size_t off = 0;
    auto carve = [&](size_t bytes) { size_t o = off; off += (bytes + 255) & ~(size_t)255; return (void*)(ws + o); };
    float*  degs   = (float*) carve((size_t)N * 4);
    float*  nrm    = (float*) carve((size_t)N * 4);
    float*  h      = (float*) carve((size_t)N * H * 4);
    __bf16* hbf    = (__bf16*)carve((size_t)N * H * 2);
    float*  hx     = (float*) carve((size_t)N * H * 4);
    float*  agg    = (float*) carve((size_t)N * H * 4);
    __bf16* wbf    = (__bf16*)carve((size_t)L * H * H * 2);
    float*  stats  = (float*) carve(2 * H * 4);
    float*  colsum = (float*) carve(H * 4);
    (void)ws_size; (void)n_in; (void)out_size;

    const int NH = N * H;
    const float invN = 1.0f / (float)N;

    // degrees + norm
    init_degs_k<<<(N + 255) / 256, 256, 0, stream>>>(degs, N);
    deg_count_k<<<(E + 255) / 256, 256, 0, stream>>>(src, degs, E);
    norm_k<<<(N + 255) / 256, 256, 0, stream>>>(degs, nrm, N);

    // h0 = node_emb[x]; bf16 copies of all layer weights
    h_init_k<<<(NH + 255) / 256, 256, 0, stream>>>(x, node_emb, h, NH);
    f32_to_bf16_k<<<(L * H * H + 255) / 256, 256, 0, stream>>>(lin_w, wbf, L * H * H);

    for (int i = 0; i < L; ++i) {
        f32_to_bf16_k<<<(NH + 255) / 256, 256, 0, stream>>>(h, hbf, NH);
        node_linear_wmma_k<<<N / 32, 256, 0, stream>>>(hbf, wbf + (size_t)i * H * H,
                                                       lin_b + i * H, hx);
        hipMemsetAsync(agg, 0, (size_t)NH * 4, stream);
        edge_fused_k<<<E / 2, 256, 0, stream>>>(ex, src, dst, nrm, hx,
                                                edge_w + (size_t)i * 7 * H,
                                                edge_b + i * H, agg);
        hipMemsetAsync(stats, 0, 2 * H * 4, stream);
        combine_stats_k<<<(N + 127) / 128, 128, 0, stream>>>(agg, hx, root_emb + i * H,
                                                             degs, h, stats, N);
        bn_apply_k<<<(NH + 255) / 256, 256, 0, stream>>>(h, stats, bn_g + i * H,
                                                         bn_b + i * H, (i < L - 1) ? 1 : 0,
                                                         NH, invN);
    }

    hipMemsetAsync(colsum, 0, H * 4, stream);
    pool_sum_k<<<(N + 127) / 128, 128, 0, stream>>>(h, colsum, N);
    out_proj_k<<<1, 128, 0, stream>>>(colsum, out_w, out_b, out, invN);
}
